// ScaleDotProductAttention_9895604650139
// MI455X (gfx1250) — compile-verified, matrix operand order checked by reading, not run on hardware
//
#include <hip/hip_runtime.h>
#include <hip/hip_bf16.h>

typedef __attribute__((ext_vector_type(16))) _Float16 v16h;
typedef __attribute__((ext_vector_type(8)))  float    v8f;

#define S_LEN   2048
#define D_HEAD  64
#define NTILES  (S_LEN / 16)     // 128 column tiles of the score matrix
#define WAVES   8
#define BLOCK   (WAVES * 32)
#define NEG_INF_SCALE (-1.0e9f)

// Per-lane K/D index mapping for the wave32 16-bit A/B fragment layout
// (ISA 7.12.2): lanes 0-15 hold K in {0..7, 16..23}, lanes 16-31 hold
// K in {8..15, 24..31}.
__device__ __forceinline__ int frag_kmap(int lane, int j) {
    int off = (lane < 16) ? 0 : 8;
    return (j < 8) ? (off + j) : (16 + off + (j - 8));
}

__global__ void ScaleDotProductAttention_9895604650139_kernel(
        const float* __restrict__ Q,
        const float* __restrict__ K,
        const float* __restrict__ V,
        const int*   __restrict__ mask,
        float*       __restrict__ out,
        float*       __restrict__ weights)
{
    extern __shared__ float smem[];
    float* logits = smem;                        // 16 x 2048 logit/prob panel
    float* oacc   = smem + 16 * S_LEN;           // 16 x 64 output accumulator
    float* red    = oacc + 16 * D_HEAD;          // 256 reduction scratch

    const int tid  = threadIdx.x;
    const int lane = tid & 31;
    const int wave = tid >> 5;
    const int mrow = lane & 15;                  // fragment row / C column
    const int bh   = blockIdx.x / NTILES;        // 0..31  (b*16 + h)
    const int qt   = blockIdx.x % NTILES;        // query tile in sequence
    const int bidx = bh >> 4;                    // batch index (H = 16)
    const float scale = 0.125f;                  // 1/sqrt(64)

    // Zero the cross-wave output accumulator (covered by phase-1 barrier).
    for (int i = tid; i < 16 * D_HEAD; i += BLOCK) oacc[i] = 0.0f;

    // ---------------- load Q fragments (16x64, two 16x32 K-chunks) --------
    const float* qrow = Q + ((size_t)bh * S_LEN + (size_t)qt * 16 + mrow) * D_HEAD;
    v16h aq0, aq1;
    #pragma unroll
    for (int j = 0; j < 16; ++j) {
        int d = frag_kmap(lane, j);
        aq0[j] = (_Float16)qrow[d];
        aq1[j] = (_Float16)qrow[d + 32];
    }

    // ---------------- phase 1: logits = scale * Q K^T + mask --------------
    const float* kbase = K + (size_t)bh * S_LEN * D_HEAD;
    for (int kt = wave; kt < NTILES; kt += WAVES) {
        const float* krow = kbase + ((size_t)kt * 16 + mrow) * D_HEAD;
        if (kt + WAVES < NTILES) {
            __builtin_prefetch(kbase + ((size_t)(kt + WAVES) * 16 + mrow) * D_HEAD, 0, 0);
        }
        v16h bk0, bk1;
        #pragma unroll
        for (int j = 0; j < 16; ++j) {
            int d = frag_kmap(lane, j);
            bk0[j] = (_Float16)krow[d];
            bk1[j] = (_Float16)krow[d + 32];
        }
        v8f acc = {};
        acc = __builtin_amdgcn_wmma_f32_16x16x32_f16(false, aq0, false, bk0,
                                                     (short)0, acc, false, false);
        acc = __builtin_amdgcn_wmma_f32_16x16x32_f16(false, aq1, false, bk1,
                                                     (short)0, acc, false, false);
        const int col   = kt * 16 + mrow;
        const float madd = (float)mask[bidx * S_LEN + col] * NEG_INF_SCALE;
        const int rbase = (lane < 16) ? 0 : 8;
        #pragma unroll
        for (int e = 0; e < 8; ++e) {
            logits[(rbase + e) * S_LEN + col] = acc[e] * scale + madd;
        }
    }
    __syncthreads();

    // ---------------- phase 2: softmax over LDS panel ---------------------
    const int r  = tid >> 4;     // row 0..15
    const int c0 = tid & 15;     // this thread covers columns c0 + 16*j
    float m = -3.0e38f;
    for (int j = 0; j < NTILES; ++j)
        m = fmaxf(m, logits[r * S_LEN + c0 + 16 * j]);
    red[tid] = m;
    __syncthreads();
    float rowmax = -3.0e38f;
    #pragma unroll
    for (int i = 0; i < 16; ++i) rowmax = fmaxf(rowmax, red[r * 16 + i]);

    float s = 0.0f;
    for (int j = 0; j < NTILES; ++j) {
        int c = c0 + 16 * j;
        float e = __expf(logits[r * S_LEN + c] - rowmax);
        logits[r * S_LEN + c] = e;
        s += e;
    }
    __syncthreads();
    red[tid] = s;
    __syncthreads();
    float tot = 0.0f;
    #pragma unroll
    for (int i = 0; i < 16; ++i) tot += red[r * 16 + i];
    const float inv = 1.0f / tot;

    float* wrow = weights + ((size_t)bh * S_LEN + (size_t)qt * 16 + r) * S_LEN;
    for (int j = 0; j < NTILES; ++j) {
        int c = c0 + 16 * j;
        float p = logits[r * S_LEN + c] * inv;
        logits[r * S_LEN + c] = p;     // keep normalized P for P*V
        wrow[c] = p;                   // the dominant HBM write, done once
    }
    __syncthreads();

    // ---------------- phase 3: out = P * V --------------------------------
    const float* vbase = V + (size_t)bh * S_LEN * D_HEAD;
    v8f co[4];
    #pragma unroll
    for (int n = 0; n < 4; ++n) { v8f z = {}; co[n] = z; }

    for (int kc = wave * 8; kc < wave * 8 + 8; ++kc) {   // 8 of 64 K-chunks
        v16h ap;
        #pragma unroll
        for (int j = 0; j < 16; ++j) {
            int k = kc * 32 + frag_kmap(lane, j);
            ap[j] = (_Float16)logits[mrow * S_LEN + k];
        }
        #pragma unroll
        for (int n = 0; n < 4; ++n) {
            v16h bv;
            #pragma unroll
            for (int j = 0; j < 16; ++j) {
                int k = kc * 32 + frag_kmap(lane, j);
                bv[j] = (_Float16)vbase[(size_t)k * D_HEAD + n * 16 + mrow];
            }
            co[n] = __builtin_amdgcn_wmma_f32_16x16x32_f16(false, ap, false, bv,
                                                           (short)0, co[n],
                                                           false, false);
        }
    }

    // Cross-wave reduction of the 16x64 output tile via LDS float atomics.
    {
        const int rbase = (lane < 16) ? 0 : 8;
        #pragma unroll
        for (int n = 0; n < 4; ++n)
            #pragma unroll
            for (int e = 0; e < 8; ++e)
                atomicAdd(&oacc[(rbase + e) * D_HEAD + n * 16 + mrow], co[n][e]);
    }
    __syncthreads();

    float* obase = out + ((size_t)bh * S_LEN + (size_t)qt * 16) * D_HEAD;
    for (int i = tid; i < 16 * D_HEAD; i += BLOCK) obase[i] = oacc[i];
}

extern "C" void kernel_launch(void* const* d_in, const int* in_sizes, int n_in,
                              void* d_out, int out_size, void* d_ws, size_t ws_size,
                              hipStream_t stream) {
    (void)in_sizes; (void)n_in; (void)out_size; (void)d_ws; (void)ws_size;
    const float* Q    = (const float*)d_in[0];
    const float* K    = (const float*)d_in[1];
    const float* V    = (const float*)d_in[2];
    const int*   mask = (const int*)d_in[3];

    float* out     = (float*)d_out;                       // (B,H,S,D) first
    float* weights = out + (size_t)2 * 16 * S_LEN * D_HEAD; // then (B,H,S,S)

    const size_t smem = (size_t)(16 * S_LEN + 16 * D_HEAD + BLOCK) * sizeof(float);
    dim3 grid(2 * 16 * NTILES);   // one block per 16-row query tile per (b,h)
    dim3 block(BLOCK);
    ScaleDotProductAttention_9895604650139_kernel<<<grid, block, smem, stream>>>(
        Q, K, V, mask, out, weights);
}
